// Decoder_11278584119887
// MI455X (gfx1250) — matrix operand order repaired and checked by reading, not compiled, not run on hardware
//
#include <hip/hip_runtime.h>

// Problem constants (match reference)
#define DD 6
#define BB 4
#define LL 2048
#define EE 1024
#define HH 16
#define FFN 4096
#define DHH 64

typedef __attribute__((ext_vector_type(16))) __bf16 v16bf;
typedef __attribute__((ext_vector_type(8)))  __bf16 v8bf;
typedef __attribute__((ext_vector_type(8)))  float  v8f;

static __device__ __forceinline__ unsigned short f2bf(float f) {
  unsigned int u = __builtin_bit_cast(unsigned int, f);
  u += 0x7fffu + ((u >> 16) & 1u);   // round-to-nearest-even
  return (unsigned short)(u >> 16);
}

// A-matrix fragment (16x32 bf16): lane r=lane&15 is row M, kh=lane>>4 selects
// K-half. Elements e: K = e + 16*kh -> one contiguous 32B run (matches B).
static __device__ __forceinline__ v16bf ld_fragA(const unsigned short* p) {
  v8bf lo = *reinterpret_cast<const v8bf*>(p);
  v8bf hi = *reinterpret_cast<const v8bf*>(p + 8);
  v16bf v;
#pragma unroll
  for (int i = 0; i < 8; ++i) { v[i] = lo[i]; v[i + 8] = hi[i]; }
  return v;
}

// B-matrix fragment (32x16 bf16): lane n=lane&15 is column N, kh=lane>>4;
// elements e: K = e + 16*kh -> one contiguous 32B run.
static __device__ __forceinline__ v16bf ld_fragB(const unsigned short* p) {
  v8bf lo = *reinterpret_cast<const v8bf*>(p);
  v8bf hi = *reinterpret_cast<const v8bf*>(p + 8);
  v16bf v;
#pragma unroll
  for (int i = 0; i < 8; ++i) { v[i] = lo[i]; v[i + 8] = hi[i]; }
  return v;
}

static __device__ __forceinline__ v8f wmma_bf16(v16bf a, v16bf b, v8f c) {
  return __builtin_amdgcn_wmma_f32_16x16x32_bf16(false, a, false, b,
                                                 (short)0, c, false, false);
}

// Async global->LDS 16B copy (CDNA5 ASYNC engine, tracked by ASYNCcnt).
// VDST = VGPR with LDS byte address (low 32 bits of the flat shared address),
// VADDR = 64-bit global address, no SADDR.
static __device__ __forceinline__ void async_cp16(const unsigned short* lds,
                                                  const unsigned short* g) {
  const unsigned loff = (unsigned)(size_t)lds;
  asm volatile("global_load_async_to_lds_b128 %0, %1, off"
               :: "v"(loff), "v"(g) : "memory");
}

// Cooperative stage of A(128x32) and B(256x32) bf16 tiles into padded LDS
// (row stride 40 elements = 80B for bank spread). 6 async instructions / wave.
static __device__ __forceinline__ void stage_tiles(
    unsigned short* sA, unsigned short* sB,
    const unsigned short* Ab, int lda, int tm0,
    const unsigned short* Bb, int ldb, int tn0, int k0, int tid) {
#pragma unroll
  for (int i = 0; i < 2; ++i) {            // 512 chunks of 16B
    const int c = tid + i * 256;
    const int row = c >> 2, k8 = (c & 3) * 8;
    async_cp16(sA + row * 40 + k8, Ab + (long)(tm0 + row) * lda + k0 + k8);
  }
#pragma unroll
  for (int i = 0; i < 4; ++i) {            // 1024 chunks of 16B
    const int c = tid + i * 256;
    const int row = c >> 2, k8 = (c & 3) * 8;
    async_cp16(sB + row * 40 + k8, Bb + (long)(tn0 + row) * ldb + k0 + k8);
  }
}

// ---------------------------------------------------------------------------
// Batched GEMM: C[z] = A[z] (MxK bf16 row-major) * B[z]^T (NxK bf16) + bias,
// optional ReLU, f32/bf16 out. Block tile 128x256 (8 waves as 2x4), each wave
// a 64x64 register tile (16 WMMAs per K-step over 8 shared fragments).
// A/B tiles double-buffered in LDS via async global->LDS + s_wait_asynccnt.
// Requires M%128==0, N%256==0, K%32==0 (true for all uses here).
// ---------------------------------------------------------------------------
template <int OUT_BF16, int RELU, int HAS_BIAS>
__global__ __launch_bounds__(256) void gemm_wmma_kernel(
    const unsigned short* __restrict__ A, long sAb, int lda,
    const unsigned short* __restrict__ Bm, long sBb, int ldb,
    const float* __restrict__ bias,
    void* __restrict__ C, long sCb, int ldc,
    int M, int N, int K) {
  __shared__ __align__(16) unsigned short smA[2][128 * 40];
  __shared__ __align__(16) unsigned short smB[2][256 * 40];
  const int tid = threadIdx.x;
  const int wave = tid >> 5, lane = tid & 31;
  const int wm = wave >> 2, wn = wave & 3;
  const int tm0 = blockIdx.y * 128;
  const int tn0 = blockIdx.x * 256;
  const unsigned short* Ab = A + (long)blockIdx.z * sAb;
  const unsigned short* Bb = Bm + (long)blockIdx.z * sBb;
  const int r = lane & 15, kh = lane >> 4;
  (void)M;

  const v8f vzero = {0.f, 0.f, 0.f, 0.f, 0.f, 0.f, 0.f, 0.f};
  v8f acc[4][4];
#pragma unroll
  for (int i = 0; i < 4; ++i)
#pragma unroll
    for (int j = 0; j < 4; ++j) acc[i][j] = vzero;

  const int nk = K >> 5;
  stage_tiles(smA[0], smB[0], Ab, lda, tm0, Bb, ldb, tn0, 0, tid);
  for (int ks = 0; ks < nk; ++ks) {
    if (ks + 1 < nk) {
      stage_tiles(smA[(ks + 1) & 1], smB[(ks + 1) & 1], Ab, lda, tm0,
                  Bb, ldb, tn0, (ks + 1) << 5, tid);
      asm volatile("s_wait_asynccnt 6" ::: "memory");  // older stage done
    } else {
      asm volatile("s_wait_asynccnt 0" ::: "memory");  // drain last stage
    }
    __syncthreads();
    const unsigned short* sa_ = smA[ks & 1];
    const unsigned short* sb_ = smB[ks & 1];
    v16bf af[4], bf[4];
#pragma unroll
    for (int i = 0; i < 4; ++i)
      af[i] = ld_fragA(sa_ + (wm * 64 + 16 * i + r) * 40 + 16 * kh);
#pragma unroll
    for (int j = 0; j < 4; ++j)
      bf[j] = ld_fragB(sb_ + (wn * 64 + 16 * j + r) * 40 + 16 * kh);
#pragma unroll
    for (int i = 0; i < 4; ++i)
#pragma unroll
      for (int j = 0; j < 4; ++j)
        acc[i][j] = wmma_bf16(af[i], bf[j], acc[i][j]);
    __syncthreads();   // all waves done reading before buffer is re-staged
  }

  // Epilogue: C layout per ISA — lane n = lane&15, VGPR e -> row e + 8*kh.
#pragma unroll
  for (int j = 0; j < 4; ++j) {
    const int n = tn0 + wn * 64 + 16 * j + r;
    const float bv = HAS_BIAS ? bias[n] : 0.f;
#pragma unroll
    for (int i = 0; i < 4; ++i) {
      const int mb = tm0 + wm * 64 + 16 * i + 8 * kh;
      if (OUT_BF16) {
        unsigned short* Cp = (unsigned short*)C + (long)blockIdx.z * sCb;
#pragma unroll
        for (int e = 0; e < 8; ++e) {
          float v = acc[i][j][e] + bv;
          if (RELU) v = fmaxf(v, 0.f);
          Cp[(long)(mb + e) * ldc + n] = f2bf(v);
        }
      } else {
        float* Cp = (float*)C + (long)blockIdx.z * sCb;
#pragma unroll
        for (int e = 0; e < 8; ++e) {
          float v = acc[i][j][e] + bv;
          if (RELU) v = fmaxf(v, 0.f);
          Cp[(long)(mb + e) * ldc + n] = v;
        }
      }
    }
  }
}

// ---------------------------------------------------------------------------
// Fused attention-weights kernel. One block = (batch b, 16 query rows).
// Per head: S = Q_h K_h^T / 8 via WMMA into LDS (16x2048 f32), row softmax,
// accumulate head mean; then rel-bias + second softmax -> bf16 P.
// Dynamic LDS = 256 KB (fits 320 KB WGP LDS; per-head scores never hit HBM).
// ---------------------------------------------------------------------------
__global__ __launch_bounds__(256) void attn_weights_kernel(
    const unsigned short* __restrict__ Q, const unsigned short* __restrict__ Kp,
    unsigned short* __restrict__ P) {
  extern __shared__ float sm[];
  float* S = sm;                 // [16][LL]
  float* ACC = sm + 16 * LL;     // [16][LL]
  const int tid = threadIdx.x;
  const int wave = tid >> 5, lane = tid & 31;
  const int b = blockIdx.y;
  const int tm = blockIdx.x * 16;
  const int r = lane & 15, kh = lane >> 4;
  const int row = tid >> 4, sub = tid & 15;   // 16 threads per score row

  for (int i = tid; i < 16 * LL; i += 256) ACC[i] = 0.f;
  __syncthreads();

  for (int h = 0; h < HH; ++h) {
    const unsigned short* qrow =
        Q + ((long)b * LL + tm + r) * EE + h * DHH + 16 * kh;
    const v16bf a0 = ld_fragA(qrow);
    const v16bf a1 = ld_fragA(qrow + 32);
    for (int nt = wave; nt < LL / 16; nt += 8) {   // uniform trip count
      const unsigned short* krow =
          Kp + ((long)b * LL + nt * 16 + r) * EE + h * DHH + 16 * kh;
      v8f acc = {0.f, 0.f, 0.f, 0.f, 0.f, 0.f, 0.f, 0.f};
      acc = wmma_bf16(a0, ld_fragB(krow), acc);
      acc = wmma_bf16(a1, ld_fragB(krow + 32), acc);
      const int n = nt * 16 + r;
#pragma unroll
      for (int i = 0; i < 8; ++i)
        S[(i + 8 * kh) * LL + n] = acc[i] * 0.125f;  // 1/sqrt(64)
    }
    __syncthreads();
    float mx = -3.0e38f;
    for (int c = sub; c < LL; c += 16) mx = fmaxf(mx, S[row * LL + c]);
#pragma unroll
    for (int off = 8; off; off >>= 1) mx = fmaxf(mx, __shfl_xor(mx, off, 16));
    float sum = 0.f;
    for (int c = sub; c < LL; c += 16) {
      float e = __expf(S[row * LL + c] - mx);
      S[row * LL + c] = e;
      sum += e;
    }
#pragma unroll
    for (int off = 8; off; off >>= 1) sum += __shfl_xor(sum, off, 16);
    const float sc = 1.f / (sum * (float)HH);
    for (int c = sub; c < LL; c += 16) ACC[row * LL + c] += S[row * LL + c] * sc;
    __syncthreads();
  }

  // second softmax with exponential relative-position bias (CSCALE = 1)
  const int gi = tm + row;
  float mx = -3.0e38f;
  for (int c = sub; c < LL; c += 16) {
    float v = ACC[row * LL + c] +
              __expf(-fabsf((float)(gi - c)) * (1.f / 16.f));
    ACC[row * LL + c] = v;
    mx = fmaxf(mx, v);
  }
#pragma unroll
  for (int off = 8; off; off >>= 1) mx = fmaxf(mx, __shfl_xor(mx, off, 16));
  float sum = 0.f;
  for (int c = sub; c < LL; c += 16) {
    float e = __expf(ACC[row * LL + c] - mx);
    ACC[row * LL + c] = e;
    sum += e;
  }
#pragma unroll
  for (int off = 8; off; off >>= 1) sum += __shfl_xor(sum, off, 16);
  const float inv = 1.f / sum;
  unsigned short* Pr = P + ((long)b * LL + gi) * LL;
  for (int c = sub; c < LL; c += 16) Pr[c] = f2bf(ACC[row * LL + c] * inv);
}

// ---------------------------------------------------------------------------
// Residual add + LayerNorm over E=1024; optional bf16 mirror for next GEMM.
// ---------------------------------------------------------------------------
__global__ __launch_bounds__(256) void add_ln_kernel(
    const float* __restrict__ X, const float* __restrict__ Y,
    const float* __restrict__ w, const float* __restrict__ b,
    float* __restrict__ outF, unsigned short* __restrict__ outB) {
  const long rowbase = (long)blockIdx.x * EE;
  const int tid = threadIdx.x;
  float4 xv = reinterpret_cast<const float4*>(X + rowbase)[tid];
  if (Y) {
    const float4 yv = reinterpret_cast<const float4*>(Y + rowbase)[tid];
    xv.x += yv.x; xv.y += yv.y; xv.z += yv.z; xv.w += yv.w;
  }
  float s = xv.x + xv.y + xv.z + xv.w;
  float ss = xv.x * xv.x + xv.y * xv.y + xv.z * xv.z + xv.w * xv.w;
  const int lane = tid & 31, wave = tid >> 5;
#pragma unroll
  for (int off = 16; off; off >>= 1) {
    s += __shfl_xor(s, off, 32);
    ss += __shfl_xor(ss, off, 32);
  }
  __shared__ float redS[8], redQ[8];
  if (lane == 0) { redS[wave] = s; redQ[wave] = ss; }
  __syncthreads();
  float ts = 0.f, tq = 0.f;
#pragma unroll
  for (int i = 0; i < 8; ++i) { ts += redS[i]; tq += redQ[i]; }
  const float mean = ts * (1.f / EE);
  const float var = tq * (1.f / EE) - mean * mean;
  const float rstd = rsqrtf(var + 1e-5f);
  const float4 wv = reinterpret_cast<const float4*>(w)[tid];
  const float4 bv = reinterpret_cast<const float4*>(b)[tid];
  float4 o;
  o.x = (xv.x - mean) * rstd * wv.x + bv.x;
  o.y = (xv.y - mean) * rstd * wv.y + bv.y;
  o.z = (xv.z - mean) * rstd * wv.z + bv.z;
  o.w = (xv.w - mean) * rstd * wv.w + bv.w;
  reinterpret_cast<float4*>(outF + rowbase)[tid] = o;
  if (outB) {
    unsigned short* p = outB + rowbase + (long)tid * 4;
    p[0] = f2bf(o.x); p[1] = f2bf(o.y); p[2] = f2bf(o.z); p[3] = f2bf(o.w);
  }
}

__global__ void f32_to_bf16_kernel(const float* __restrict__ in,
                                   unsigned short* __restrict__ out, long n) {
  long i = (long)blockIdx.x * blockDim.x + threadIdx.x;
  const long stride = (long)gridDim.x * blockDim.x;
  for (; i < n; i += stride) out[i] = f2bf(in[i]);
}

// bf16 [rows,cols] -> [cols,rows], 32x32 LDS tiles, grid.z = batch
__global__ __launch_bounds__(256) void transpose_bf16_kernel(
    const unsigned short* __restrict__ in, unsigned short* __restrict__ out,
    int rows, int cols) {
  __shared__ unsigned short tile[32][33];
  const long boff = (long)blockIdx.z * rows * cols;
  const int c0 = blockIdx.x * 32, r0 = blockIdx.y * 32;
  const int tx = threadIdx.x & 31, ty = threadIdx.x >> 5;
  for (int i = ty; i < 32; i += 8)
    tile[i][tx] = in[boff + (long)(r0 + i) * cols + c0 + tx];
  __syncthreads();
  for (int i = ty; i < 32; i += 8)
    out[boff + (long)(c0 + i) * rows + r0 + tx] = tile[tx][i];
}

// ---------------------------------------------------------------------------
extern "C" void kernel_launch(void* const* d_in, const int* in_sizes, int n_in,
                              void* d_out, int out_size, void* d_ws,
                              size_t ws_size, hipStream_t stream) {
  (void)in_sizes; (void)n_in; (void)out_size; (void)ws_size;
  const float* x    = (const float*)d_in[0];
  const float* Wq   = (const float*)d_in[1];
  const float* Wk   = (const float*)d_in[2];
  const float* bq   = (const float*)d_in[3];
  const float* bk   = (const float*)d_in[4];
  const float* ln1w = (const float*)d_in[5];
  const float* ln1b = (const float*)d_in[6];
  const float* ln2w = (const float*)d_in[7];
  const float* ln2b = (const float*)d_in[8];
  const float* ln3w = (const float*)d_in[9];
  const float* ln3b = (const float*)d_in[10];
  const float* W1   = (const float*)d_in[11];
  const float* b1   = (const float*)d_in[12];
  const float* W2   = (const float*)d_in[13];
  const float* b2   = (const float*)d_in[14];

  char* p = (char*)d_ws;
  auto carve = [&](size_t bytes) -> char* {
    char* q = p;
    p += (bytes + 255) & ~(size_t)255;
    return q;
  };
  const size_t nBLE = (size_t)BB * LL * EE;
  unsigned short* wq_bf = (unsigned short*)carve((size_t)DD * EE * EE * 2);
  unsigned short* wk_bf = (unsigned short*)carve((size_t)DD * EE * EE * 2);
  unsigned short* w1_bf = (unsigned short*)carve((size_t)DD * FFN * EE * 2);
  unsigned short* w2_bf = (unsigned short*)carve((size_t)DD * EE * FFN * 2);
  unsigned short* hb    = (unsigned short*)carve(nBLE * 2);           // h, bf16
  unsigned short* hbT   = (unsigned short*)carve(nBLE * 2);           // [B,E,L]
  unsigned short* qb    = (unsigned short*)carve(nBLE * 2);
  unsigned short* kb    = (unsigned short*)carve(nBLE * 2);
  unsigned short* h2b   = (unsigned short*)carve(nBLE * 2);
  unsigned short* f1b   = (unsigned short*)carve((size_t)BB * LL * FFN * 2);
  unsigned short* Pb    = (unsigned short*)carve((size_t)BB * LL * LL * 2);
  float* sa   = (float*)carve(nBLE * 4);
  float* h1   = (float*)carve(nBLE * 4);
  float* h2   = (float*)carve(nBLE * 4);
  float* ff   = (float*)carve(nBLE * 4);
  float* hbuf = (float*)carve(nBLE * 4);

  const dim3 blk(256, 1, 1);
  const int Mrows = BB * LL;  // 8192

  // Per-launch bf16 weight conversion + initial stream conversion
  f32_to_bf16_kernel<<<1024, 256, 0, stream>>>(Wq, wq_bf, (long)DD * EE * EE);
  f32_to_bf16_kernel<<<1024, 256, 0, stream>>>(Wk, wk_bf, (long)DD * EE * EE);
  f32_to_bf16_kernel<<<2048, 256, 0, stream>>>(W1, w1_bf, (long)DD * FFN * EE);
  f32_to_bf16_kernel<<<2048, 256, 0, stream>>>(W2, w2_bf, (long)DD * EE * FFN);
  f32_to_bf16_kernel<<<1024, 256, 0, stream>>>(x, hb, (long)nBLE);

  const float* hcur = x;
  for (int d = 0; d < DD; ++d) {
    // hbT = transpose(hb) per batch (B operand of the P @ h GEMM)
    transpose_bf16_kernel<<<dim3(EE / 32, LL / 32, BB), blk, 0, stream>>>(
        hb, hbT, LL, EE);

    // Q = hb Wq^T + bq ; K = hb Wk^T + bk   (bf16 outputs)
    gemm_wmma_kernel<1, 0, 1><<<dim3(EE / 256, Mrows / 128, 1), blk, 0, stream>>>(
        hb, 0, EE, wq_bf + (size_t)d * EE * EE, 0, EE, bq + (size_t)d * EE,
        qb, 0, EE, Mrows, EE, EE);
    gemm_wmma_kernel<1, 0, 1><<<dim3(EE / 256, Mrows / 128, 1), blk, 0, stream>>>(
        hb, 0, EE, wk_bf + (size_t)d * EE * EE, 0, EE, bk + (size_t)d * EE,
        kb, 0, EE, Mrows, EE, EE);

    // P = softmax( mean_h softmax(QK^T/sqrt(d)) + rel )  -> bf16 [B,L,L]
    attn_weights_kernel<<<dim3(LL / 16, BB, 1), blk,
                          2u * 16u * LL * sizeof(float), stream>>>(qb, kb, Pb);

    // sa = P @ h   (batched over B; B-operand = h^T)
    gemm_wmma_kernel<0, 0, 0><<<dim3(EE / 256, LL / 128, BB), blk, 0, stream>>>(
        Pb, (long)LL * LL, LL, hbT, (long)EE * LL, LL, nullptr,
        sa, (long)LL * EE, EE, LL, EE, LL);

    // h1 = LN(h + sa); h2 = LN(h1) (+ bf16 mirror for FFN)
    add_ln_kernel<<<dim3(Mrows), blk, 0, stream>>>(
        sa, hcur, ln1w + (size_t)d * EE, ln1b + (size_t)d * EE, h1, nullptr);
    add_ln_kernel<<<dim3(Mrows), blk, 0, stream>>>(
        h1, nullptr, ln2w + (size_t)d * EE, ln2b + (size_t)d * EE, h2, h2b);

    // FFN: f1 = relu(h2 W1^T + b1) bf16 ; ff = f1 W2^T + b2 f32
    gemm_wmma_kernel<1, 1, 1><<<dim3(FFN / 256, Mrows / 128, 1), blk, 0, stream>>>(
        h2b, 0, EE, w1_bf + (size_t)d * FFN * EE, 0, EE, b1 + (size_t)d * FFN,
        f1b, 0, FFN, Mrows, FFN, EE);
    gemm_wmma_kernel<0, 0, 1><<<dim3(EE / 256, Mrows / 128, 1), blk, 0, stream>>>(
        f1b, 0, FFN, w2_bf + (size_t)d * EE * FFN, 0, FFN, b2 + (size_t)d * EE,
        ff, 0, EE, Mrows, EE, FFN);

    // h = LN(h2 + ff) (+ bf16 mirror for next layer); last layer -> d_out
    float* outF = (d == DD - 1) ? (float*)d_out : hbuf;
    add_ln_kernel<<<dim3(Mrows), blk, 0, stream>>>(
        ff, h2, ln3w + (size_t)d * EE, ln3b + (size_t)d * EE, outF, hb);
    hcur = hbuf;
  }
}
